// LSTM_5471788335583
// MI455X (gfx1250) — compile-verified, matrix operand order checked by reading, not run on hardware
//
#include <hip/hip_runtime.h>
#include <math.h>

// ---------- types ----------
typedef __bf16 bf16x8  __attribute__((ext_vector_type(8)));
typedef __bf16 bf16x16 __attribute__((ext_vector_type(16)));
typedef float  f32x8   __attribute__((ext_vector_type(8)));

union BF16Frag { bf16x16 v; bf16x8 h[2]; };

__device__ inline unsigned short f2bf(float f) {
    unsigned u = __builtin_bit_cast(unsigned, f);
    unsigned r = u + 0x7FFFu + ((u >> 16) & 1u);   // round-to-nearest-even
    return (unsigned short)(r >> 16);
}

__device__ inline float sigmoidf_(float x) { return 1.f / (1.f + __expf(-x)); }

// Load a 16x32 bf16 fragment (A layout; B uses the mirrored identical pattern)
// src is row-major [rows][ldk] bf16. Lane<16: row=lane, K 0..7 & 16..23;
// lane>=16: row=lane-16, K 8..15 & 24..31.  (ISA 7.12.2 16-bit A layout)
// Works for both global and LDS-derived pointers (addrspace inferred).
__device__ inline bf16x16 load_frag(const unsigned short* src, int row0, int k0, int ldk) {
    int lane = threadIdx.x & 31;
    int hi   = lane >> 4;
    int r    = lane & 15;
    const unsigned short* p = src + (size_t)(row0 + r) * ldk + k0 + hi * 8;
    BF16Frag f;
    f.h[0] = *(const bf16x8*)(p);
    f.h[1] = *(const bf16x8*)(p + 16);
    return f.v;
}

__device__ inline f32x8 wmma_bf16(bf16x16 a, bf16x16 b, f32x8 c) {
    return __builtin_amdgcn_wmma_f32_16x16x32_bf16(false, a, false, b, (short)0, c, false, false);
}

// Monotonic grid barrier across NBLK resident workgroups.
__device__ inline void grid_barrier(unsigned* cnt, unsigned nblk, unsigned phase) {
    __syncthreads();
    if (threadIdx.x == 0) {
        __threadfence();
        atomicAdd(cnt, 1u);
        unsigned target = nblk * (phase + 1u);
        while (__atomic_load_n(cnt, __ATOMIC_RELAXED) < target) {
            __builtin_amdgcn_s_sleep(1);
        }
        __threadfence();
    }
    __syncthreads();
}

// ---------- weight packing ----------
// Packed row rp = blk*(4*Hs) + g*Hs + j  <->  original row g*H + blk*Hs + j
__global__ void pack_lstm_kernel(const float* Wih, const float* Whh,
                                 const float* bih, const float* bhh,
                                 unsigned short* Wihp, unsigned short* Whhp,
                                 float* biasP, float* W1p,
                                 int H, int Hs, int I, int isL1) {
    int FH = 4 * H;
    int rp = blockIdx.x * blockDim.x + threadIdx.x;
    if (rp >= FH) return;
    int GW = 4 * Hs;
    int blk = rp / GW, rem = rp % GW;
    int g = rem / Hs, j = rem % Hs;
    int orig = g * H + blk * Hs + j;
    for (int k = 0; k < H; ++k) Whhp[(size_t)rp * H + k] = f2bf(Whh[(size_t)orig * H + k]);
    if (!isL1)
        for (int k = 0; k < I; ++k) Wihp[(size_t)rp * I + k] = f2bf(Wih[(size_t)orig * I + k]);
    biasP[rp] = bih[orig] + bhh[orig];
    if (isL1) W1p[rp] = Wih[orig];     // I == 1
}

__global__ void pack_fc_kernel(const float* Wfc1, unsigned short* Wp, int n) {
    int i = blockIdx.x * blockDim.x + threadIdx.x;
    if (i < n) Wp[i] = f2bf(Wfc1[i]);
}

__global__ void init_counters_kernel(unsigned* c) {
    if (threadIdx.x < 64) c[threadIdx.x] = 0u;
}

// ---------- persistent LSTM layer ----------
// H: hidden, Hs: per-block slice of H, I: input width (0 => layer1 scalar-x path)
// 512 threads = 16 waves (4 per SIMD32) for latency hiding on the sequential loop.
template <int H, int Hs, int I, bool L1>
__global__ void lstm_layer_kernel(const float* x1,              // [T,B,1]   (L1 only)
                                  const unsigned short* Xprev,  // [T,B,I]   bf16
                                  const unsigned short* Wihp,   // [4H,I]    bf16 packed
                                  const unsigned short* Whhp,   // [4H,H]    bf16 packed
                                  const float* biasP,           // [4H]      packed
                                  const float* W1p,             // [4H]      packed (L1)
                                  const float* h0, const float* c0,  // [B,H]
                                  unsigned short* Hbuf,         // [2,B,H]   bf16 ping-pong
                                  float* cbuf,                  // [B,H]
                                  unsigned short* Yout,         // [T,B,H]   bf16
                                  float* hT, float* cT,         // [B,H] outputs
                                  unsigned* bar) {
    constexpr int B = 256, T = 512;
    constexpr int NBLK = H / Hs;
    constexpr int NT = (4 * Hs) / 16;   // N-tiles per block
    constexpr int GW = 4 * Hs;          // gate-slice width
    constexpr int JS = Hs / 2;          // columns per thread in elementwise phase

    // dynamic LDS: [gates f32 B*GW][WhhL bf16 GW*H][WihL bf16 GW*I][biasL f32 GW][w1L f32 GW]
    extern __shared__ char smem[];
    float*          gsm   = (float*)smem;
    unsigned short* WhhL  = (unsigned short*)(smem + (size_t)B * GW * 4);
    unsigned short* WihL  = WhhL + (size_t)GW * H;
    float*          biasL = (float*)(WihL + (size_t)GW * I);
    float*          w1L   = biasL + GW;

    const int tid  = threadIdx.x;
    const int lane = tid & 31;
    const int wave = tid >> 5;          // 16 waves, each owns one 16-row M-tile
    const int blk  = blockIdx.x;
    const int nb0  = blk * GW;          // packed weight/bias row base
    const int c0s  = blk * Hs;          // h-column base

    // stage this block's weight rows into LDS (time-invariant; removes L2 traffic
    // from the per-step critical path)
    {
        const uint4* Wg = (const uint4*)(Whhp + (size_t)nb0 * H);
        uint4*       Wl = (uint4*)WhhL;
        for (int idx = tid; idx < GW * H / 8; idx += blockDim.x) Wl[idx] = Wg[idx];
        if (I > 0) {
            const uint4* Xg = (const uint4*)(Wihp + (size_t)nb0 * I);
            uint4*       Xl = (uint4*)WihL;
            for (int idx = tid; idx < GW * I / 8; idx += blockDim.x) Xl[idx] = Xg[idx];
        }
        for (int idx = tid; idx < GW; idx += blockDim.x) {
            biasL[idx] = biasP[nb0 + idx];
            if (L1) w1L[idx] = W1p[nb0 + idx];
        }
    }

    // init h (bf16 ping buffer 0) and c slice
    for (int idx = tid; idx < B * Hs; idx += blockDim.x) {
        int b = idx / Hs, j = idx % Hs, col = c0s + j;
        Hbuf[(size_t)b * H + col] = f2bf(h0[(size_t)b * H + col]);
        cbuf[(size_t)b * H + col] = c0[(size_t)b * H + col];
    }
    unsigned ph = 0;
    grid_barrier(bar, NBLK, ph++);      // includes __syncthreads (covers LDS staging)

    for (int t = 0; t < T; ++t) {
        const unsigned short* Hr = Hbuf + (size_t)(t & 1) * B * H;
        unsigned short*       Hw = Hbuf + (size_t)((t + 1) & 1) * B * H;

        f32x8 acc[NT];
        #pragma unroll
        for (int n = 0; n < NT; ++n)
            #pragma unroll
            for (int r = 0; r < 8; ++r) acc[n][r] = 0.f;

        if (I > 0) {  // input projection  x_t @ Wih^T
            const unsigned short* Xt = Xprev + (size_t)t * B * I;
            #pragma unroll
            for (int kt = 0; kt < I / 32; ++kt) {
                bf16x16 a = load_frag(Xt, wave * 16, kt * 32, I);
                #pragma unroll
                for (int n = 0; n < NT; ++n) {
                    bf16x16 bb = load_frag(WihL, n * 16, kt * 32, I);
                    acc[n] = wmma_bf16(a, bb, acc[n]);
                }
            }
        }
        // recurrent projection  h @ Whh^T
        #pragma unroll
        for (int kt = 0; kt < H / 32; ++kt) {
            bf16x16 a = load_frag(Hr, wave * 16, kt * 32, H);
            #pragma unroll
            for (int n = 0; n < NT; ++n) {
                bf16x16 bb = load_frag(WhhL, n * 16, kt * 32, H);
                acc[n] = wmma_bf16(a, bb, acc[n]);
            }
        }

        // accumulators -> LDS gates  (C layout: reg r, lane -> m=r+8*(lane>>4), n=lane&15)
        {
            int hi = lane >> 4, r15 = lane & 15;
            #pragma unroll
            for (int n = 0; n < NT; ++n)
                #pragma unroll
                for (int r = 0; r < 8; ++r) {
                    int m = wave * 16 + hi * 8 + r;
                    gsm[m * GW + n * 16 + r15] = acc[n][r];
                }
        }
        __syncthreads();

        // elementwise phase: 2 threads per batch row, each handles JS columns
        {
            int b  = tid & 255;
            int j0 = (tid >> 8) * JS;
            float xv = L1 ? x1[(size_t)t * B + b] : 0.f;
            #pragma unroll
            for (int jj = 0; jj < JS; ++jj) {
                int j = j0 + jj;
                float gi = gsm[b * GW + 0 * Hs + j] + biasL[0 * Hs + j];
                float gf = gsm[b * GW + 1 * Hs + j] + biasL[1 * Hs + j];
                float gg = gsm[b * GW + 2 * Hs + j] + biasL[2 * Hs + j];
                float go = gsm[b * GW + 3 * Hs + j] + biasL[3 * Hs + j];
                if (L1) {
                    gi += xv * w1L[0 * Hs + j];
                    gf += xv * w1L[1 * Hs + j];
                    gg += xv * w1L[2 * Hs + j];
                    go += xv * w1L[3 * Hs + j];
                }
                float i_ = sigmoidf_(gi), f_ = sigmoidf_(gf), o_ = sigmoidf_(go);
                float g_ = tanhf(gg);
                int col = c0s + j;
                float c = f_ * cbuf[(size_t)b * H + col] + i_ * g_;
                cbuf[(size_t)b * H + col] = c;
                float h = o_ * tanhf(c);
                unsigned short hb = f2bf(h);
                Hw[(size_t)b * H + col] = hb;
                Yout[((size_t)t * B + b) * H + col] = hb;
                if (t == T - 1) {
                    hT[(size_t)b * H + col] = h;
                    cT[(size_t)b * H + col] = c;
                }
            }
        }
        grid_barrier(bar, NBLK, ph++);   // h(t+1) visible device-wide
    }
}

// ---------- fused FC1(ReLU)+FC2 ----------
__global__ void fc_kernel(const unsigned short* Y3,   // [T*B, 256] bf16
                          const unsigned short* Wg,   // [128, 256] bf16
                          const float* b1, const float* w2, const float* b2,
                          float* y) {                 // [T*B]
    constexpr int K = 256, N = 128, NT = 8;
    const int tid = threadIdx.x, lane = tid & 31, wave = tid >> 5;
    const int m0 = blockIdx.x * 128 + wave * 16;

    extern __shared__ char smem[];
    unsigned short* WL = (unsigned short*)smem;     // [128][256]
    {
        const uint4* src = (const uint4*)Wg;
        uint4*       dst = (uint4*)WL;
        for (int idx = tid; idx < N * K / 8; idx += blockDim.x) dst[idx] = src[idx];
    }
    __syncthreads();

    f32x8 acc[NT];
    #pragma unroll
    for (int n = 0; n < NT; ++n)
        #pragma unroll
        for (int r = 0; r < 8; ++r) acc[n][r] = 0.f;

    #pragma unroll
    for (int kt = 0; kt < K / 32; ++kt) {
        bf16x16 a = load_frag(Y3, m0, kt * 32, K);
        #pragma unroll
        for (int n = 0; n < NT; ++n) {
            bf16x16 bb = load_frag(WL, n * 16, kt * 32, K);
            acc[n] = wmma_bf16(a, bb, acc[n]);
        }
    }

    int hi = lane >> 4, r15 = lane & 15;
    float part[8];
    #pragma unroll
    for (int r = 0; r < 8; ++r) part[r] = 0.f;
    #pragma unroll
    for (int n = 0; n < NT; ++n) {
        int col = n * 16 + r15;
        float bb1 = b1[col], ww2 = w2[col];
        #pragma unroll
        for (int r = 0; r < 8; ++r) {
            float v = acc[n][r] + bb1;
            v = v > 0.f ? v : 0.f;
            part[r] += v * ww2;
        }
    }
    // reduce across the 16 lanes of each half (columns of one row)
    #pragma unroll
    for (int r = 0; r < 8; ++r)
        #pragma unroll
        for (int mask = 1; mask < 16; mask <<= 1)
            part[r] += __shfl_xor(part[r], mask, 32);
    if (r15 == 0) {
        float bb2 = b2[0];
        #pragma unroll
        for (int r = 0; r < 8; ++r)
            y[(size_t)m0 + hi * 8 + r] = part[r] + bb2;
    }
}

// ---------- host ----------
extern "C" void kernel_launch(void* const* d_in, const int* in_sizes, int n_in,
                              void* d_out, int out_size, void* d_ws, size_t ws_size,
                              hipStream_t stream) {
    const float* x    = (const float*)d_in[0];
    const float* h1   = (const float*)d_in[1];
    const float* c1   = (const float*)d_in[2];
    const float* h2   = (const float*)d_in[3];
    const float* c2   = (const float*)d_in[4];
    const float* h3   = (const float*)d_in[5];
    const float* c3   = (const float*)d_in[6];
    const float* Wih1 = (const float*)d_in[7];
    const float* Whh1 = (const float*)d_in[8];
    const float* bih1 = (const float*)d_in[9];
    const float* bhh1 = (const float*)d_in[10];
    const float* Wih2 = (const float*)d_in[11];
    const float* Whh2 = (const float*)d_in[12];
    const float* bih2 = (const float*)d_in[13];
    const float* bhh2 = (const float*)d_in[14];
    const float* Wih3 = (const float*)d_in[15];
    const float* Whh3 = (const float*)d_in[16];
    const float* bih3 = (const float*)d_in[17];
    const float* bhh3 = (const float*)d_in[18];
    const float* Wfc1 = (const float*)d_in[19];
    const float* bfc1 = (const float*)d_in[20];
    const float* Wfc2 = (const float*)d_in[21];
    const float* bfc2 = (const float*)d_in[22];

    float* out = (float*)d_out;
    float* y   = out;                 // [512*256]
    float* h1n = y   + 131072;        // [256*64]
    float* h2n = h1n + 16384;         // [256*128]
    float* h3n = h2n + 32768;         // [256*256]
    float* c1n = h3n + 65536;
    float* c2n = c1n + 16384;
    float* c3n = c2n + 32768;

    // bump allocator over d_ws
    char* ws = (char*)d_ws;
    size_t off = 0;
    auto alloc = [&](size_t bytes) -> void* {
        void* p = ws + off;
        off = (off + bytes + 255) & ~(size_t)255;
        return p;
    };
    unsigned*       bar   = (unsigned*)alloc(256);
    unsigned short* Y1    = (unsigned short*)alloc((size_t)512 * 256 * 64 * 2);
    unsigned short* Y2    = (unsigned short*)alloc((size_t)512 * 256 * 128 * 2);
    unsigned short* Y3    = (unsigned short*)alloc((size_t)512 * 256 * 256 * 2);
    unsigned short* Hb1   = (unsigned short*)alloc((size_t)2 * 256 * 64 * 2);
    unsigned short* Hb2   = (unsigned short*)alloc((size_t)2 * 256 * 128 * 2);
    unsigned short* Hb3   = (unsigned short*)alloc((size_t)2 * 256 * 256 * 2);
    float*          cb1   = (float*)alloc((size_t)256 * 64 * 4);
    float*          cb2   = (float*)alloc((size_t)256 * 128 * 4);
    float*          cb3   = (float*)alloc((size_t)256 * 256 * 4);
    unsigned short* Whhp1 = (unsigned short*)alloc((size_t)256 * 64 * 2);
    unsigned short* Whhp2 = (unsigned short*)alloc((size_t)512 * 128 * 2);
    unsigned short* Whhp3 = (unsigned short*)alloc((size_t)1024 * 256 * 2);
    unsigned short* Wihp2 = (unsigned short*)alloc((size_t)512 * 64 * 2);
    unsigned short* Wihp3 = (unsigned short*)alloc((size_t)1024 * 128 * 2);
    float*          biasP1 = (float*)alloc(256 * 4);
    float*          biasP2 = (float*)alloc(512 * 4);
    float*          biasP3 = (float*)alloc(1024 * 4);
    float*          W1p    = (float*)alloc(256 * 4);
    unsigned short* Wfc1p  = (unsigned short*)alloc((size_t)128 * 256 * 2);

    init_counters_kernel<<<1, 64, 0, stream>>>(bar);

    pack_lstm_kernel<<<(256 + 63) / 64, 64, 0, stream>>>(
        Wih1, Whh1, bih1, bhh1, Whhp1 /*unused*/, Whhp1, biasP1, W1p, 64, 16, 1, 1);
    pack_lstm_kernel<<<(512 + 63) / 64, 64, 0, stream>>>(
        Wih2, Whh2, bih2, bhh2, Wihp2, Whhp2, biasP2, W1p /*unused*/, 128, 16, 64, 0);
    pack_lstm_kernel<<<(1024 + 63) / 64, 64, 0, stream>>>(
        Wih3, Whh3, bih3, bhh3, Wihp3, Whhp3, biasP3, W1p /*unused*/, 256, 32, 128, 0);
    pack_fc_kernel<<<(128 * 256 + 255) / 256, 256, 0, stream>>>(Wfc1, Wfc1p, 128 * 256);

    // dynamic LDS: gates + staged weights + bias vectors
    const size_t lds1 = (size_t)256 * 64 * 4  + (size_t)64 * 64 * 2                          + 64 * 8;
    const size_t lds2 = (size_t)256 * 64 * 4  + (size_t)64 * 128 * 2 + (size_t)64 * 64 * 2   + 64 * 8;
    const size_t lds3 = (size_t)256 * 128 * 4 + (size_t)128 * 256 * 2 + (size_t)128 * 128 * 2 + 128 * 8;

    // persistent per-layer kernels (grid = NBLK co-resident blocks, 16 waves each)
    lstm_layer_kernel<64, 16, 0, true><<<4, 512, lds1, stream>>>(
        x, nullptr, Whhp1, Whhp1, biasP1, W1p, h1, c1, Hb1, cb1, Y1, h1n, c1n, bar + 0);
    lstm_layer_kernel<128, 16, 64, false><<<8, 512, lds2, stream>>>(
        nullptr, Y1, Wihp2, Whhp2, biasP2, nullptr, h2, c2, Hb2, cb2, Y2, h2n, c2n, bar + 16);
    lstm_layer_kernel<256, 32, 128, false><<<8, 512, lds3, stream>>>(
        nullptr, Y2, Wihp3, Whhp3, biasP3, nullptr, h3, c3, Hb3, cb3, Y3, h3n, c3n, bar + 32);

    fc_kernel<<<1024, 256, 65536, stream>>>(Y3, Wfc1p, bfc1, Wfc2, bfc2, y);
}
